// RelationalConvGraphLayer_40029095198875
// MI455X (gfx1250) — matrix operand order repaired and checked by reading, not compiled, not run on hardware
//
#include <hip/hip_runtime.h>
#include <hip/hip_bf16.h>

// RGCN layer, fused: out[b,n,u] = relu( sum_r (adj[b,r] @ feat[b]) @ ker[r] + sum_r bias[r] )
// B=64 R=5 N=512 A=64 U=128. Memory-bound on the single 335MB adjacency stream.

typedef __attribute__((ext_vector_type(16))) __bf16 v16bf;
typedef __attribute__((ext_vector_type(8)))  __bf16 v8bf;
typedef __attribute__((ext_vector_type(4)))  __bf16 v4bf;
typedef __attribute__((ext_vector_type(8)))  float  v8f;
typedef __attribute__((ext_vector_type(4)))  float  vf4;

#define B_   64
#define R_   5
#define N_   512
#define A_   64
#define U_   128
#define FT_STRIDE 520   // padded featT row stride (bf16 elems) to soften bank conflicts

// Native f32 -> bf16 conversion: let clang emit hardware cvt (pairs -> v_cvt_pk_bf16_f32)
__device__ __forceinline__ __bf16 f2bf(float f) { return (__bf16)f; }

__device__ __forceinline__ v4bf cvt4(vf4 f) {
    return __builtin_convertvector(f, v4bf);
}
__device__ __forceinline__ v8bf concat4(v4bf a, v4bf b) {
    return __builtin_shufflevector(a, b, 0,1,2,3,4,5,6,7);
}
__device__ __forceinline__ v16bf concat8(v8bf lo, v8bf hi) {
    return __builtin_shufflevector(lo, hi, 0,1,2,3,4,5,6,7,8,9,10,11,12,13,14,15);
}

__global__ __launch_bounds__(256, 1)
void rgcn_fused_wmma(const float* __restrict__ adj,
                     const float* __restrict__ feat,
                     const float* __restrict__ ker,
                     const float* __restrict__ bias,
                     float* __restrict__ out)
{
    extern __shared__ char smem[];
    __bf16* featT = (__bf16*)smem;                                      // [A_=64][FT_STRIDE]
    __bf16* kerT  = (__bf16*)(smem + A_ * FT_STRIDE * 2);               // [U_=128][A_=64]
    __bf16* aggS  = (__bf16*)(smem + A_ * FT_STRIDE * 2 + U_ * A_ * 2); // 8 waves * [16][64]

    const int tid  = threadIdx.x;
    const int lane = tid & 31;
    const int wave = tid >> 5;
    const int lc   = lane & 15;      // column / row-in-tile index
    const int hi   = lane >> 4;      // 0 or 1: K-chunk half selector

    const int b     = blockIdx.x >> 2;            // 64 batches
    const int tileN = (blockIdx.x & 3) * 128;     // 4 N-tiles of 128 rows
    const int row0  = tileN + wave * 16;          // this wave's 16 N-rows

    // ---- stage features[b]^T into LDS as bf16: featT[a][m] ----
    const float* featB = feat + (size_t)b * N_ * A_;
    for (int i = tid; i < N_ * A_; i += 256) {
        int m = i >> 6, a = i & 63;               // feat[m][a], a fastest (coalesced read)
        featT[a * FT_STRIDE + m] = f2bf(featB[i]);
    }

    __bf16* aggW = aggS + wave * (16 * 64);       // wave-private agg scratch [16][64]

    const v8f vzero = {0.f,0.f,0.f,0.f,0.f,0.f,0.f,0.f};
    v8f acc[8];
    #pragma unroll
    for (int t = 0; t < 8; ++t) acc[t] = vzero;

    for (int r = 0; r < R_; ++r) {
        __syncthreads();   // featT ready (r==0) / prior-iteration kerT reads done

        // ---- stage ker[r]^T into LDS as bf16: kerT[u][a] ----
        const float* kerR = ker + (size_t)r * A_ * U_;
        for (int i = tid; i < A_ * U_; i += 256) {
            int a = i >> 7, u = i & 127;          // ker[a][u], u fastest (coalesced read)
            kerT[u * A_ + a] = f2bf(kerR[i]);
        }
        __syncthreads();

        // ---- stage 1: agg(16x64) = adj_tile(16x512) @ feat(512x64), bf16 WMMA ----
        v8f aggc[4];
        #pragma unroll
        for (int t = 0; t < 4; ++t) aggc[t] = vzero;

        // lane's adjacency row (A-matrix rows: lanes 0-15 and 16-31 both map M=lc)
        const float* adjRow = adj + (((size_t)(b * R_ + r) * N_ + row0 + lc) * N_);

        for (int kb = 0; kb < 16; ++kb) {
            // A-fragment: K-chunks [k0..k0+7] and [k0+16..k0+23], k0 = kb*32 + hi*8
            const int k0 = kb * 32 + (hi << 3);
            // One-shot 335MB stream: non-temporal so L2 keeps the reused feat/ker instead
            vf4 f0 = __builtin_nontemporal_load((const vf4*)(adjRow + k0));
            vf4 f1 = __builtin_nontemporal_load((const vf4*)(adjRow + k0) + 1);
            vf4 f2 = __builtin_nontemporal_load((const vf4*)(adjRow + k0 + 16));
            vf4 f3 = __builtin_nontemporal_load((const vf4*)(adjRow + k0 + 16) + 1);
            if (kb < 15) __builtin_prefetch(adjRow + k0 + 32, 0, 1); // global_prefetch_b8

            v16bf afrag = concat8(concat4(cvt4(f0), cvt4(f1)),
                                  concat4(cvt4(f2), cvt4(f3)));

            #pragma unroll
            for (int ab = 0; ab < 4; ++ab) {
                // B-fragment: column a = ab*16+lc, K chunks [k0..k0+7] and [k0+16..k0+23], k0 = kb*32 + hi*8
                const __bf16* bp = featT + (ab * 16 + lc) * FT_STRIDE + kb * 32 + (hi << 3);
                v8bf blo = *(const v8bf*)bp;
                v8bf bhi = *(const v8bf*)(bp + 16);
                v16bf bfrag = concat8(blo, bhi);
                aggc[ab] = __builtin_amdgcn_wmma_f32_16x16x32_bf16(
                    false, afrag, false, bfrag, (short)0, aggc[ab], false, false);
            }
        }

        // ---- spill agg tiles (D-layout) to wave-private LDS, row-major bf16 [16][64] ----
        #pragma unroll
        for (int ab = 0; ab < 4; ++ab) {
            #pragma unroll
            for (int v = 0; v < 8; ++v) {
                int m = v + (hi << 3);            // D-layout: vgpr v -> row v (+8 for hi lanes)
                aggW[m * 64 + ab * 16 + lc] = f2bf(aggc[ab][v]);
            }
        }
        // same-wave DS ops are in-order; compiler inserts s_wait_dscnt for the reload below

        // ---- stage 2: acc(16x128) += agg(16x64) @ ker_r(64x128) ----
        #pragma unroll
        for (int kb = 0; kb < 2; ++kb) {
            const __bf16* ap = aggW + lc * 64 + kb * 32 + (hi << 3);
            v8bf a0 = *(const v8bf*)ap;           // K chunk [k0..k0+7]
            v8bf a1 = *(const v8bf*)(ap + 16);    // K chunk [k0+16..k0+23]
            v16bf afrag2 = concat8(a0, a1);
            #pragma unroll
            for (int ub = 0; ub < 8; ++ub) {
                const __bf16* bp = kerT + (ub * 16 + lc) * A_ + kb * 32 + (hi << 3);
                v8bf blo = *(const v8bf*)bp;
                v8bf bhi = *(const v8bf*)(bp + 16);
                v16bf bfrag = concat8(blo, bhi);
                acc[ub] = __builtin_amdgcn_wmma_f32_16x16x32_bf16(
                    false, afrag2, false, bfrag, (short)0, acc[ub], false, false);
            }
        }
    }

    // ---- epilogue: + sum_r bias, ReLU, non-temporal store (write-once output) ----
    #pragma unroll
    for (int ub = 0; ub < 8; ++ub) {
        int u = ub * 16 + lc;
        float bs = 0.f;
        #pragma unroll
        for (int r = 0; r < R_; ++r) bs += bias[r * U_ + u];
        #pragma unroll
        for (int v = 0; v < 8; ++v) {
            int n = row0 + v + (hi << 3);
            float val = acc[ub][v] + bs;
            val = val > 0.f ? val : 0.f;
            __builtin_nontemporal_store(val, &out[((size_t)b * N_ + n) * U_ + u]);
        }
    }
}

extern "C" void kernel_launch(void* const* d_in, const int* in_sizes, int n_in,
                              void* d_out, int out_size, void* d_ws, size_t ws_size,
                              hipStream_t stream) {
    (void)in_sizes; (void)n_in; (void)d_ws; (void)ws_size; (void)out_size;
    const float* adj  = (const float*)d_in[0];   // [64,5,512,512] f32
    const float* feat = (const float*)d_in[1];   // [64,512,64]    f32
    const float* ker  = (const float*)d_in[2];   // [5,64,128]     f32
    const float* bias = (const float*)d_in[3];   // [5,1,128]      f32
    float* out = (float*)d_out;                  // [64,512,128]   f32

    const size_t shmem = (size_t)A_ * FT_STRIDE * 2   // featT  ~66.6 KB
                       + (size_t)U_ * A_ * 2          // kerT    16 KB
                       + (size_t)8 * 16 * 64 * 2;     // aggS    16 KB
    dim3 grid(B_ * (N_ / 128));                  // 256 workgroups
    dim3 block(256);                             // 8 wave32 per WG
    rgcn_fused_wmma<<<grid, block, shmem, stream>>>(adj, feat, ker, bias, out);
}
